// BiasedAttentionTransformerBlock_81612968558724
// MI455X (gfx1250) — compile-verified, matrix operand order checked by reading, not run on hardware
//
#include <hip/hip_runtime.h>
#include <hip/hip_bf16.h>
#include <stdint.h>

// Problem dims (compile-time)
#define B_   128
#define L_   128
#define E_   512
#define H_   8
#define DH_  64
#define KR_  16
#define F_   2048
#define BL_  (B_ * L_)

typedef __bf16 v4bf  __attribute__((ext_vector_type(4)));
typedef __bf16 v8bf  __attribute__((ext_vector_type(8)));
typedef __bf16 v16bf __attribute__((ext_vector_type(16)));
typedef float  v8f   __attribute__((ext_vector_type(8)));
typedef int    vint4 __attribute__((__vector_size__(16)));

// TDM descriptor group types (6-arg builtin on this toolchain)
typedef unsigned int tdm_v4u __attribute__((__vector_size__(16)));
typedef int          tdm_v8i __attribute__((__vector_size__(32)));
typedef int          tdm_v4i __attribute__((__vector_size__(16)));

#if defined(__gfx1250__) && \
    __has_builtin(__builtin_amdgcn_global_load_async_to_lds_b128) && \
    __has_builtin(__builtin_amdgcn_s_wait_asynccnt)
#define HAVE_ASYNC_LDS 1
#else
#define HAVE_ASYNC_LDS 0
#endif

#if defined(__gfx1250__) && \
    __has_builtin(__builtin_amdgcn_tensor_load_to_lds) && \
    __has_builtin(__builtin_amdgcn_s_wait_tensorcnt)
#define HAVE_TDM 1
#else
#define HAVE_TDM 0
#endif

// ---------------------------------------------------------------------------
// WMMA helpers (CDNA5 16x16x32 bf16 -> f32)
// ---------------------------------------------------------------------------
__device__ __forceinline__ v16bf frag_cat(v8bf lo, v8bf hi) {
    return __builtin_shufflevector(lo, hi, 0, 1, 2, 3, 4, 5, 6, 7,
                                   8, 9, 10, 11, 12, 13, 14, 15);
}

// Load a 16x32 bf16 A/B fragment from a row-major matrix.
// Lane l<16 holds row l, K in {0..7, 16..23}; lane l+16 holds row l-16,
// K in {8..15, 24..31}  (matches the CDNA5 16-bit A/B VGPR layout, ISA 7.12.2).
// Works for generic pointers (global or LDS).
__device__ __forceinline__ v16bf load_frag(const __bf16* __restrict__ base,
                                           int ld, int lane) {
    int r  = lane & 15;
    int hf = lane >> 4;  // 0: K 0..7 / 16..23   1: K 8..15 / 24..31
    const __bf16* p = base + (size_t)r * ld + hf * 8;
    v8bf lo = *(const v8bf*)p;          // K  base+0..7
    v8bf hi = *(const v8bf*)(p + 16);   // K  base+16..23
    return frag_cat(lo, hi);
}

__device__ __forceinline__ v8f wmma_bf16(v16bf a, v16bf b, v8f c) {
    return __builtin_amdgcn_wmma_f32_16x16x32_bf16(
        /*neg_a=*/false, a, /*neg_b=*/false, b,
        /*c_mod=*/(short)0, c, /*reuse_a=*/false, /*reuse_b=*/false);
}

#if HAVE_TDM
// Issue a 2D TDM load of a (tile_h x tile_w) bf16 tile from a row-major
// matrix (row stride = tensor_w elements) into LDS at lds_off.
// D# layout per CDNA5 ISA 8.3/8.4; groups 2/3 (and the trailing group) zero
// for a 2D tensor.
__device__ __forceinline__ void tdm_load_2d_bf16(const __bf16* gsrc,
                                                 unsigned lds_off,
                                                 int tensor_w, int tensor_h,
                                                 int tile_w, int tile_h) {
    unsigned long long ga = (unsigned long long)(size_t)gsrc;
    tdm_v4u g0;
    g0[0] = 1u;                                   // count=1, user descriptor
    g0[1] = lds_off;                              // LDS byte address
    g0[2] = (unsigned)(ga & 0xFFFFFFFFu);         // global_addr[31:0]
    g0[3] = (unsigned)((ga >> 32) & 0x01FFFFFFu)  // global_addr[56:32]
          | (2u << 30);                           // type=2 ("image")
    tdm_v8i g1;
    g1[0] = (1 << 16);                            // data_size=1 (2 bytes)
    g1[1] = (tensor_w & 0xFFFF) << 16;            // tensor_dim0[15:0] @ bits 63:48
    g1[2] = ((tensor_w >> 16) & 0xFFFF)           // tensor_dim0[31:16]
          | ((tensor_h & 0xFFFF) << 16);          // tensor_dim1[15:0]
    g1[3] = ((tensor_h >> 16) & 0xFFFF)           // tensor_dim1[31:16]
          | ((tile_w & 0xFFFF) << 16);            // tile_dim0
    g1[4] = (tile_h & 0xFFFF);                    // tile_dim1 (tile_dim2=0)
    g1[5] = tensor_w;                             // tensor_dim0_stride[31:0]
    g1[6] = 0;                                    // stride[47:32] | dim1_stride lo
    g1[7] = 0;
    tdm_v4i z4 = {0, 0, 0, 0};
    tdm_v8i z8 = {0, 0, 0, 0, 0, 0, 0, 0};
    __builtin_amdgcn_tensor_load_to_lds(g0, g1, z4, z4, z8, 0);
}
#endif

// ---------------------------------------------------------------------------
// f32 -> bf16 conversion (weights)
// ---------------------------------------------------------------------------
__global__ __launch_bounds__(256)
void cvt_bf16_kernel(const float* __restrict__ in, __bf16* __restrict__ out, int n) {
    int i = (blockIdx.x * 256 + threadIdx.x) * 4;
    if (i + 3 < n) {
        float4 v = *(const float4*)(in + i);
        v4bf o = { (__bf16)v.x, (__bf16)v.y, (__bf16)v.z, (__bf16)v.w };
        *(v4bf*)(out + i) = o;
    }
}

// ---------------------------------------------------------------------------
// LayerNorm over E=512 cols, one row per 128-thread block.
// Writes f32 and bf16 copies.
// ---------------------------------------------------------------------------
__global__ __launch_bounds__(128)
void layernorm_kernel(const float* __restrict__ x,
                      const float* __restrict__ g,
                      const float* __restrict__ bta,
                      float* __restrict__ yf,
                      __bf16* __restrict__ yb) {
    int row = blockIdx.x;
    int t = threadIdx.x;
    const float* xr = x + (size_t)row * E_;
    float4 v = *(const float4*)(xr + t * 4);
    float s  = v.x + v.y + v.z + v.w;
    float s2 = v.x * v.x + v.y * v.y + v.z * v.z + v.w * v.w;
#pragma unroll
    for (int o = 16; o >= 1; o >>= 1) {
        s  += __shfl_xor(s, o, 32);
        s2 += __shfl_xor(s2, o, 32);
    }
    __shared__ float ps[4], ps2[4];
    int w = t >> 5, lane = t & 31;
    if (lane == 0) { ps[w] = s; ps2[w] = s2; }
    __syncthreads();
    s  = ps[0] + ps[1] + ps[2] + ps[3];
    s2 = ps2[0] + ps2[1] + ps2[2] + ps2[3];
    float mu   = s * (1.0f / E_);
    float var  = s2 * (1.0f / E_) - mu * mu;
    float rstd = rsqrtf(fmaxf(var, 0.0f) + 1e-5f);
    float4 gv = *(const float4*)(g + t * 4);
    float4 bv = *(const float4*)(bta + t * 4);
    float4 y;
    y.x = (v.x - mu) * rstd * gv.x + bv.x;
    y.y = (v.y - mu) * rstd * gv.y + bv.y;
    y.z = (v.z - mu) * rstd * gv.z + bv.z;
    y.w = (v.w - mu) * rstd * gv.w + bv.w;
    *(float4*)(yf + (size_t)row * E_ + t * 4) = y;
    v4bf ob = { (__bf16)y.x, (__bf16)y.y, (__bf16)y.z, (__bf16)y.w };
    *(v4bf*)(yb + (size_t)row * E_ + t * 4) = ob;
}

// ---------------------------------------------------------------------------
// RBF attention bias: compute 16 Gaussians once per (b,i,j), dot with all
// 8 heads. Output layout [B,H,L,L] f32.
// ---------------------------------------------------------------------------
__global__ __launch_bounds__(256)
void rbf_bias_kernel(const float* __restrict__ d,
                     const float* __restrict__ mu,
                     const float* __restrict__ gamma,
                     const float* __restrict__ rw,       // [H,KR]
                     float* __restrict__ biasOut) {
    __shared__ float smu[KR_];
    __shared__ float srw[H_ * KR_];
    if (threadIdx.x < KR_)       smu[threadIdx.x] = mu[threadIdx.x];
    if (threadIdx.x < H_ * KR_)  srw[threadIdx.x] = rw[threadIdx.x];
    __syncthreads();
    float gmm = gamma[0];
    size_t t = (size_t)blockIdx.x * 256 + threadIdx.x;   // over B*L*L
    float dv = d[t];
    float acc[H_];
#pragma unroll
    for (int h = 0; h < H_; ++h) acc[h] = 0.0f;
#pragma unroll
    for (int k = 0; k < KR_; ++k) {
        float df = dv - smu[k];
        float f = __expf(-gmm * df * df);
#pragma unroll
        for (int h = 0; h < H_; ++h) acc[h] += srw[h * KR_ + k] * f;
    }
    size_t bb = t / (L_ * L_);
    size_t ij = t % (L_ * L_);
#pragma unroll
    for (int h = 0; h < H_; ++h)
        biasOut[(bb * H_ + h) * (size_t)(L_ * L_) + ij] = acc[h];
}

// ---------------------------------------------------------------------------
// Generic WMMA GEMM:  Y[m,n] = sum_k X[m,k] * W[n,k]  (+ epilogue)
// X: [M,K] bf16 row-major (ldx), W: [N,K] bf16 row-major (ldw).
// Block: 256 thr = 8 waves (2x4 wave grid), block tile 64x128, wave tile 32x32.
// Per K-step: A tile (64x32) staged via GLOBAL_LOAD_ASYNC_TO_LDS_B128,
//             W tile (128x32) staged via TENSOR_LOAD_TO_LDS (TDM, wave 0).
// MODE 1: outB = bf16(acc + bias)                  (QKV proj)
// MODE 2: outF = acc + bias + resid                (O proj + residual)
// MODE 3: outB = bf16(relu(acc + bias))            (FFN up)
// MODE 4: outF = (rowmask? 0 : acc+bias) + resid   (FFN down + final add)
// ---------------------------------------------------------------------------
template <int MODE>
__global__ __launch_bounds__(256)
void gemm_bf16_wmma(const __bf16* __restrict__ X, int ldx,
                    const __bf16* __restrict__ W, int ldw,
                    const float* __restrict__ bias,
                    const float* __restrict__ resid,
                    const unsigned char* __restrict__ rowmask,
                    float* __restrict__ outF,
                    __bf16* __restrict__ outB,
                    int M, int N, int K) {
    __shared__ __bf16 tileA[64 * 32];    // 4 KB: X tile
    __shared__ __bf16 tileW[128 * 32];   // 8 KB: W tile
    int w    = threadIdx.x >> 5;
    int lane = threadIdx.x & 31;
    int wm = w >> 2, wn = w & 3;
    int bm = blockIdx.y * 64;
    int bn = blockIdx.x * 128;
    int m0 = wm * 32;                 // wave row offset within block tile
    int nl = wn * 32;                 // wave col offset within W tile (local)
    int n0 = bn + nl;                 // wave col offset (global)

    v8f c00 = {0, 0, 0, 0, 0, 0, 0, 0};
    v8f c01 = {0, 0, 0, 0, 0, 0, 0, 0};
    v8f c10 = {0, 0, 0, 0, 0, 0, 0, 0};
    v8f c11 = {0, 0, 0, 0, 0, 0, 0, 0};

    int tRow = threadIdx.x >> 2;        // 0..63
    int tCol = (threadIdx.x & 3) * 8;   // 0,8,16,24 (bf16 elems, 16B chunks)
    const __bf16* gA = X + (size_t)(bm + tRow) * ldx + tCol;

#if HAVE_TDM
    unsigned ldsWOff =
        (unsigned)(size_t)(__attribute__((address_space(3))) char*)(void*)&tileW[0];
#endif

    for (int kk = 0; kk < K; kk += 32) {
        // ---- stage X tile (async DMA to LDS, one b128 per thread) ----
#if HAVE_ASYNC_LDS
        __builtin_amdgcn_global_load_async_to_lds_b128(
            (__attribute__((address_space(1))) vint4*)(void*)(gA + kk),
            (__attribute__((address_space(3))) vint4*)(void*)&tileA[tRow * 32 + tCol],
            0, 0);
        __builtin_amdgcn_s_wait_asynccnt(0);
#else
        *(float4*)&tileA[tRow * 32 + tCol] = *(const float4*)(gA + kk);
#endif
        // ---- stage W tile (Tensor Data Mover, issued once by wave 0) ----
#if HAVE_TDM
        if (threadIdx.x == 0) {
            tdm_load_2d_bf16(W + (size_t)bn * ldw + kk, ldsWOff,
                             /*tensor_w=*/ldw, /*tensor_h=*/N,
                             /*tile_w=*/32, /*tile_h=*/128);
            __builtin_amdgcn_s_wait_tensorcnt(0);
        }
#else
        for (int c = threadIdx.x; c < 512; c += 256) {
            int r = c >> 2, co = (c & 3) * 8;
            *(float4*)&tileW[r * 32 + co] =
                *(const float4*)(W + (size_t)(bn + r) * ldw + kk + co);
        }
#endif
        __syncthreads();

        v16bf a0 = load_frag(&tileA[(m0 + 0) * 32], 32, lane);
        v16bf a1 = load_frag(&tileA[(m0 + 16) * 32], 32, lane);
        v16bf b0 = load_frag(&tileW[(nl + 0) * 32], 32, lane);
        v16bf b1 = load_frag(&tileW[(nl + 16) * 32], 32, lane);

        c00 = wmma_bf16(a0, b0, c00);
        c01 = wmma_bf16(a0, b1, c01);
        c10 = wmma_bf16(a1, b0, c10);
        c11 = wmma_bf16(a1, b1, c11);
        __syncthreads();
    }

    // Epilogue: C fragment layout — VGPR r, lane<16: (M=r, N=lane);
    // lane>=16: (M=r+8, N=lane-16).
    int cr  = lane & 15;
    int chh = lane >> 4;
    v8f cc[2][2] = {{c00, c01}, {c10, c11}};
#pragma unroll
    for (int i = 0; i < 2; ++i) {
#pragma unroll
        for (int j = 0; j < 2; ++j) {
            v8f c = cc[i][j];
            int colb = n0 + j * 16 + cr;
            float bb = bias[colb];
#pragma unroll
            for (int r = 0; r < 8; ++r) {
                int row = bm + m0 + i * 16 + r + chh * 8;
                size_t off = (size_t)row * N + colb;
                float v = c[r] + bb;
                if (MODE == 1) {
                    outB[off] = (__bf16)v;
                } else if (MODE == 2) {
                    outF[off] = v + resid[off];
                } else if (MODE == 3) {
                    outB[off] = (__bf16)fmaxf(v, 0.0f);
                } else { // MODE == 4
                    if (rowmask[row]) v = 0.0f;
                    outF[off] = v + resid[off];
                }
            }
        }
    }
}

// ---------------------------------------------------------------------------
// Fused biased attention for one (b,h): scores = qk^T/8 + bias, softmax,
// o = attn @ v. Block = 256 thr (8 waves); wave w owns query rows 16w..16w+15.
// ---------------------------------------------------------------------------
__global__ __launch_bounds__(256)
void attention_kernel(const __bf16* __restrict__ qb,
                      const __bf16* __restrict__ kb,
                      const __bf16* __restrict__ vb,
                      const float* __restrict__ bias,         // [B,H,L,L]
                      const unsigned char* __restrict__ amask, // [B,1,L,L]
                      __bf16* __restrict__ ob) {
    int bh = blockIdx.x;
    int b = bh / H_;
    int h = bh % H_;
    int w = threadIdx.x >> 5;
    int lane = threadIdx.x & 31;

    __shared__ __bf16 vT[DH_ * L_];          // v transposed: [dh][token], 16 KB
    __shared__ __bf16 probs[8 * 16 * L_];    // per-wave 16x128 probs,  32 KB

    // Stage v^T cooperatively so PV B-fragments are contiguous row loads.
    const __bf16* vbase = vb + (size_t)b * L_ * E_ + h * DH_;
    for (int idx = threadIdx.x; idx < L_ * DH_; idx += 256) {
        int k = idx >> 6;    // token 0..127
        int n = idx & 63;    // dh 0..63
        vT[n * L_ + k] = vbase[(size_t)k * E_ + n];
    }
    __syncthreads();

    int qi0 = w * 16;
    const __bf16* qbase = qb + ((size_t)(b * L_ + qi0)) * E_ + h * DH_;
    v16bf aq0 = load_frag(qbase, E_, lane);       // K = dh 0..31
    v16bf aq1 = load_frag(qbase + 32, E_, lane);  // K = dh 32..63

    const __bf16* kbase = kb + (size_t)(b * L_) * E_ + h * DH_;
    v8f sc[8];
#pragma unroll
    for (int f = 0; f < 8; ++f) {
        const __bf16* kf = kbase + (size_t)(f * 16) * E_;
        v16bf b0 = load_frag(kf, E_, lane);
        v16bf b1 = load_frag(kf + 32, E_, lane);
        v8f c = {0, 0, 0, 0, 0, 0, 0, 0};
        c = wmma_bf16(aq0, b0, c);
        c = wmma_bf16(aq1, b1, c);
        sc[f] = c;
    }

    // scale + bias + mask
    const float scale = 0.125f;   // 1/sqrt(64)
    int cr = lane & 15, ch = lane >> 4;
    const float* bbase = bias + ((size_t)b * H_ + h) * (size_t)(L_ * L_);
    const unsigned char* mbase = amask + (size_t)b * L_ * L_;
#pragma unroll
    for (int f = 0; f < 8; ++f) {
#pragma unroll
        for (int r = 0; r < 8; ++r) {
            int row = qi0 + r + ch * 8;
            int col = f * 16 + cr;
            size_t o = (size_t)row * L_ + col;
            float v = sc[f][r] * scale + bbase[o];
            if (mbase[o]) v = -1e9f;
            sc[f][r] = v;
        }
    }

    // softmax per row; each 16-lane half owns a distinct row per VGPR index r
#pragma unroll
    for (int r = 0; r < 8; ++r) {
        float m = -3.4e38f;
#pragma unroll
        for (int f = 0; f < 8; ++f) m = fmaxf(m, sc[f][r]);
#pragma unroll
        for (int o = 8; o >= 1; o >>= 1) m = fmaxf(m, __shfl_xor(m, o, 32));
        float s = 0.0f;
#pragma unroll
        for (int f = 0; f < 8; ++f) {
            float e = __expf(sc[f][r] - m);
            sc[f][r] = e;
            s += e;
        }
#pragma unroll
        for (int o = 8; o >= 1; o >>= 1) s += __shfl_xor(s, o, 32);
        float inv = 1.0f / s;
#pragma unroll
        for (int f = 0; f < 8; ++f) sc[f][r] *= inv;
    }

    // Round-trip probs through wave-private LDS to re-fragment as A-matrix.
    __bf16* pw = probs + w * 16 * L_;
#pragma unroll
    for (int f = 0; f < 8; ++f) {
#pragma unroll
        for (int r = 0; r < 8; ++r) {
            int row = r + ch * 8;
            int col = f * 16 + cr;
            pw[row * L_ + col] = (__bf16)sc[f][r];
        }
    }

    // o = probs(16x128) @ v(128x64): 4 output fragments, K in 4 chunks of 32.
#pragma unroll
    for (int g = 0; g < 4; ++g) {
        v8f acc = {0, 0, 0, 0, 0, 0, 0, 0};
#pragma unroll
        for (int kc = 0; kc < 4; ++kc) {
            v16bf ap = load_frag(pw + kc * 32, L_, lane);
            v16bf bv = load_frag(vT + (g * 16) * L_ + kc * 32, L_, lane);
            acc = wmma_bf16(ap, bv, acc);
        }
        __bf16* obase = ob + ((size_t)(b * L_ + qi0)) * E_ + h * DH_ + g * 16;
#pragma unroll
        for (int r = 0; r < 8; ++r) {
            int row = r + ch * 8;
            obase[(size_t)row * E_ + cr] = (__bf16)acc[r];
        }
    }
}

// ---------------------------------------------------------------------------
// Launcher
// ---------------------------------------------------------------------------
extern "C" void kernel_launch(void* const* d_in, const int* in_sizes, int n_in,
                              void* d_out, int out_size, void* d_ws, size_t ws_size,
                              hipStream_t stream) {
    (void)in_sizes; (void)n_in; (void)out_size; (void)ws_size;
    const float* e    = (const float*)d_in[1];
    const float* dmat = (const float*)d_in[2];
    const unsigned char* amask = (const unsigned char*)d_in[3];
    const unsigned char* pmask = (const unsigned char*)d_in[4];
    const float* Wq = (const float*)d_in[5];   const float* bq = (const float*)d_in[6];
    const float* Wk = (const float*)d_in[7];   const float* bk = (const float*)d_in[8];
    const float* Wv = (const float*)d_in[9];   const float* bv = (const float*)d_in[10];
    const float* Wo = (const float*)d_in[11];  const float* bo = (const float*)d_in[12];
    const float* ln1g = (const float*)d_in[13]; const float* ln1b = (const float*)d_in[14];
    const float* ln2g = (const float*)d_in[15]; const float* ln2b = (const float*)d_in[16];
    const float* W1 = (const float*)d_in[17];  const float* b1 = (const float*)d_in[18];
    const float* W2 = (const float*)d_in[19];  const float* b2 = (const float*)d_in[20];
    const float* rmu = (const float*)d_in[21];
    const float* rg  = (const float*)d_in[22];
    const float* rw  = (const float*)d_in[23];
    float* out = (float*)d_out;

    // Bump allocator over workspace
    char* wp = (char*)d_ws;
    auto alloc = [&](size_t bytes) -> char* {
        char* p = wp;
        wp += (bytes + 255) & ~(size_t)255;
        return p;
    };
    float*  e0f = (float*) alloc((size_t)BL_ * E_ * 4);
    __bf16* e0b = (__bf16*)alloc((size_t)BL_ * E_ * 2);
    __bf16* wqb = (__bf16*)alloc((size_t)E_ * E_ * 2);
    __bf16* wkb = (__bf16*)alloc((size_t)E_ * E_ * 2);
    __bf16* wvb = (__bf16*)alloc((size_t)E_ * E_ * 2);
    __bf16* wob = (__bf16*)alloc((size_t)E_ * E_ * 2);
    __bf16* w1b = (__bf16*)alloc((size_t)F_ * E_ * 2);
    __bf16* w2b = (__bf16*)alloc((size_t)E_ * F_ * 2);
    __bf16* qbuf = (__bf16*)alloc((size_t)BL_ * E_ * 2);
    __bf16* kbuf = (__bf16*)alloc((size_t)BL_ * E_ * 2);
    __bf16* vbuf = (__bf16*)alloc((size_t)BL_ * E_ * 2);
    float*  biasbuf = (float*)alloc((size_t)B_ * H_ * L_ * L_ * 4);
    __bf16* obuf = (__bf16*)alloc((size_t)BL_ * E_ * 2);
    float*  e1r  = (float*) alloc((size_t)BL_ * E_ * 4);
    float*  e2f  = (float*) alloc((size_t)BL_ * E_ * 4);
    __bf16* e2b  = (__bf16*)alloc((size_t)BL_ * E_ * 2);
    __bf16* h1b  = (__bf16*)alloc((size_t)BL_ * F_ * 2);

    // 1. LN1: e -> e0 (f32 + bf16)
    layernorm_kernel<<<BL_, 128, 0, stream>>>(e, ln1g, ln1b, e0f, e0b);

    // 2. Weight conversions to bf16
    auto cvt = [&](const float* src, __bf16* dst, int n) {
        cvt_bf16_kernel<<<(n / 4 + 255) / 256, 256, 0, stream>>>(src, dst, n);
    };
    cvt(Wq, wqb, E_ * E_);
    cvt(Wk, wkb, E_ * E_);
    cvt(Wv, wvb, E_ * E_);
    cvt(Wo, wob, E_ * E_);
    cvt(W1, w1b, F_ * E_);
    cvt(W2, w2b, E_ * F_);

    // 3. RBF attention bias [B,H,L,L]
    rbf_bias_kernel<<<(B_ * L_ * L_) / 256, 256, 0, stream>>>(
        dmat, rmu, rg, rw, biasbuf);

    // 4. QKV projections (bf16 WMMA GEMM), outputs bf16
    dim3 gE(E_ / 128, BL_ / 64);
    gemm_bf16_wmma<1><<<gE, 256, 0, stream>>>(e0b, E_, wqb, E_, bq,
        nullptr, nullptr, nullptr, qbuf, BL_, E_, E_);
    gemm_bf16_wmma<1><<<gE, 256, 0, stream>>>(e0b, E_, wkb, E_, bk,
        nullptr, nullptr, nullptr, kbuf, BL_, E_, E_);
    gemm_bf16_wmma<1><<<gE, 256, 0, stream>>>(e0b, E_, wvb, E_, bv,
        nullptr, nullptr, nullptr, vbuf, BL_, E_, E_);

    // 5. Fused biased attention + softmax + PV
    attention_kernel<<<B_ * H_, 256, 0, stream>>>(
        qbuf, kbuf, vbuf, biasbuf, amask, obuf);

    // 6. O-projection + residual: e1r = o @ Wo^T + bo + e0
    gemm_bf16_wmma<2><<<gE, 256, 0, stream>>>(obuf, E_, wob, E_, bo,
        e0f, nullptr, e1r, nullptr, BL_, E_, E_);

    // 7. LN2: e2 = LN(e1 + e0)
    layernorm_kernel<<<BL_, 128, 0, stream>>>(e1r, ln2g, ln2b, e2f, e2b);

    // 8. FFN up: h1 = relu(e2 @ W1^T + b1), bf16
    dim3 gF(F_ / 128, BL_ / 64);
    gemm_bf16_wmma<3><<<gF, 256, 0, stream>>>(e2b, E_, w1b, E_, b1,
        nullptr, nullptr, nullptr, h1b, BL_, F_, E_);

    // 9. FFN down + mask + final residual: out = where(pad,0,h1@W2^T+b2) + e2
    gemm_bf16_wmma<4><<<gE, 256, 0, stream>>>(h1b, F_, w2b, F_, b2,
        e2f, pmask, out, nullptr, BL_, E_, F_);
}
